// SelfAttentionHead_83236466196859
// MI455X (gfx1250) — matrix operand order repaired and checked by reading, not compiled, not run on hardware
//
#include <hip/hip_runtime.h>

// ---------------------------------------------------------------------------
// Causal self-attention head for MI455X (gfx1250, wave32).
// bf16 WMMA compute, fp32 accumulate, async global->LDS double buffering.
// __launch_bounds__(128,1) relaxes the VGPR budget so fragment loads can stay
// live ahead of the WMMAs instead of being sunk into load->wait->wmma chains.
// B=4, T=2048, C=1024, H=64.
// ---------------------------------------------------------------------------

typedef __bf16 bf16;
typedef __attribute__((ext_vector_type(16))) __bf16 v16bf;
typedef __attribute__((ext_vector_type(8)))  __bf16 v8bf;
typedef __attribute__((ext_vector_type(8)))  float  v8f;
typedef __attribute__((ext_vector_type(4)))  float  v4f;

#define B_ 4
#define T_ 2048
#define C_ 1024
#define H_ 64

#define WMMA_BF16(a, b, c) \
  __builtin_amdgcn_wmma_f32_16x16x32_bf16(false, (a), false, (b), (short)0, (c), false, false)

// Load a 16-half fragment from two 8-element (16B-aligned) pieces.
static __device__ inline v16bf ld16(const bf16* p0, const bf16* p1) {
  v8bf a = *(const v8bf*)p0;
  v8bf b = *(const v8bf*)p1;
  v16bf r;
#pragma unroll
  for (int i = 0; i < 8; ++i) { r[i] = a[i]; r[i + 8] = b[i]; }
  return r;
}

// Flat LDS pointers carry the LDS byte offset in bits [31:0] (ISA 10.2).
static __device__ inline unsigned lds_off(const void* p) {
  return (unsigned)(unsigned long long)p;
}

// gfx1250 async global->LDS copy, 16B per lane, tracked by ASYNCcnt.
static __device__ inline void async_b128(unsigned lds, unsigned goff,
                                         unsigned long long sbase) {
  asm volatile("global_load_async_to_lds_b128 %0, %1, %2"
               :: "v"(lds), "v"(goff), "s"(sbase)
               : "memory");
}
static __device__ inline void wait_async0() {
  asm volatile("s_wait_asynccnt 0" ::: "memory");
}

// ---------------------------------------------------------------------------
// Kernel 1: W [C,H] fp32 -> Wt [3][H][C] bf16 (contiguous q|k|v)
// ---------------------------------------------------------------------------
__global__ void wt_prep(const float* __restrict__ Wq, const float* __restrict__ Wk,
                        const float* __restrict__ Wv, bf16* __restrict__ Wt) {
  int i = blockIdx.x * blockDim.x + threadIdx.x;   // 0 .. H*C-1
  int c = i & (C_ - 1);
  int h = i >> 10;
  Wt[(0 * H_ + h) * C_ + c] = (bf16)Wq[c * H_ + h];
  Wt[(1 * H_ + h) * C_ + c] = (bf16)Wk[c * H_ + h];
  Wt[(2 * H_ + h) * C_ + c] = (bf16)Wv[c * H_ + h];
}

// ---------------------------------------------------------------------------
// Kernel 2: fused q/k/v projection, async-LDS double-buffered W chunks.
// One wave per 16-row tile; 4 waves per block share the W chunk in LDS.
// ---------------------------------------------------------------------------
__global__ void __launch_bounds__(128, 1)
qkv_proj(const float* __restrict__ x, const bf16* __restrict__ Wt,
         bf16* __restrict__ qo, bf16* __restrict__ ko, bf16* __restrict__ vto) {
  __shared__ bf16 Wl[2][3 * 64 * 32];              // 2 x 12 KB
  const int tid  = threadIdx.x;
  const int lane = tid & 31;
  const int wave = tid >> 5;
  const int tile = blockIdx.x * 4 + wave;          // 0..511
  const int b    = tile >> 7;
  const int t0   = (tile & 127) << 4;
  const int n    = lane & 15;
  const int g    = lane >> 4;

  const unsigned long long wbase = (unsigned long long)Wt;

  auto issue = [&](int kc, int buf) {
#pragma unroll
    for (int j = 0; j < 6; ++j) {
      const int e   = j * 128 + tid;               // 0..767
      const int row = e >> 2;                      // m*64 + h, 0..191
      const int seg = e & 3;                       // 16B segment of 64B row
      async_b128(lds_off(&Wl[buf][row * 32 + seg * 8]),
                 (unsigned)((row * C_ + kc + seg * 8) * 2), wbase);
    }
  };
  issue(0, 0);

  v8f acc[3][4];
#pragma unroll
  for (int m2 = 0; m2 < 3; ++m2)
#pragma unroll
    for (int nt = 0; nt < 4; ++nt) acc[m2][nt] = (v8f){};

  const float* xr = x + (long)(b * T_ + t0 + n) * C_;

  for (int i = 0; i < C_ / 32; ++i) {
    const int kc = i * 32;
    wait_async0();
    __syncthreads();
    if (i + 1 < C_ / 32) issue(kc + 32, (i + 1) & 1);
    const bf16* wl = &Wl[i & 1][0] + (n * 32 + g * 8);   // per-lane base

    // A fragment from x (fp32 -> bf16 inline)
    v4f f0 = *(const v4f*)(xr + kc + g * 8);
    v4f f1 = *(const v4f*)(xr + kc + g * 8 + 4);
    v4f f2 = *(const v4f*)(xr + kc + 16 + g * 8);
    v4f f3 = *(const v4f*)(xr + kc + 20 + g * 8);
    v16bf a;
#pragma unroll
    for (int t = 0; t < 4; ++t) {
      a[t]      = (bf16)f0[t];
      a[t + 4]  = (bf16)f1[t];
      a[t + 8]  = (bf16)f2[t];
      a[t + 12] = (bf16)f3[t];
    }

    // Load ALL 12 B-fragments up front (24 ds_load_b128), then 12 WMMAs.
    v16bf fr[12];
#pragma unroll
    for (int f = 0; f < 12; ++f) {
      const bf16* p = wl + f * (16 * 32);          // (m2*4+nt) tiles, 1KB apart
      fr[f] = ld16(p, p + 16);
    }
#pragma unroll
    for (int f = 0; f < 12; ++f)
      acc[f >> 2][f & 3] = WMMA_BF16(a, fr[f], acc[f >> 2][f & 3]);
  }

#pragma unroll
  for (int nt = 0; nt < 4; ++nt) {
#pragma unroll
    for (int r = 0; r < 8; ++r) {
      long row = (long)(b * T_ + t0 + g * 8 + r);
      qo[row * H_ + nt * 16 + n] = (bf16)acc[0][nt][r];
      ko[row * H_ + nt * 16 + n] = (bf16)acc[1][nt][r];
    }
    v8bf pk;
#pragma unroll
    for (int r = 0; r < 8; ++r) pk[r] = (bf16)acc[2][nt][r];
    *(v8bf*)(vto + (long)(b * H_ + nt * 16 + n) * T_ + t0 + g * 8) = pk;
  }
}

// ---------------------------------------------------------------------------
// Kernel 3: flash attention, async-LDS double-buffered K/V tiles.
// Block = 64 queries (4 waves x 16); all waves share K/V staging.
// ---------------------------------------------------------------------------
__global__ void __launch_bounds__(128, 1)
attn(const bf16* __restrict__ q, const bf16* __restrict__ k,
     const bf16* __restrict__ vt, float* __restrict__ out) {
  __shared__ bf16 Kl[2][32 * 64];                  // 2 x 4 KB
  __shared__ bf16 Vl[2][64 * 32];                  // 2 x 4 KB
  __shared__ bf16 Pl[4][16 * 32];                  // per-wave P staging
  const int tid  = threadIdx.x;
  const int lane = tid & 31;
  const int wave = tid >> 5;
  const int b    = blockIdx.x >> 5;                // 32 blocks per batch
  const int q0b  = (blockIdx.x & 31) * 64;
  const int q0   = q0b + wave * 16;
  const int n    = lane & 15;
  const int g    = lane >> 4;

  const unsigned long long kbase = (unsigned long long)(k + (long)b * T_ * H_);
  const unsigned long long vbase = (unsigned long long)(vt + (long)b * H_ * T_);

  auto issue = [&](int k0, int buf) {
#pragma unroll
    for (int j = 0; j < 2; ++j) {                  // K tile: 32 x 64 bf16
      const int e = j * 128 + tid;                 // 0..255
      const int row = e >> 3, seg = e & 7;
      async_b128(lds_off(&Kl[buf][row * 64 + seg * 8]),
                 (unsigned)(((k0 + row) * H_ + seg * 8) * 2), kbase);
    }
#pragma unroll
    for (int j = 0; j < 2; ++j) {                  // V tile: 64 x 32 bf16
      const int e = j * 128 + tid;
      const int row = e >> 2, seg = e & 3;
      async_b128(lds_off(&Vl[buf][row * 32 + seg * 8]),
                 (unsigned)((row * T_ + k0 + seg * 8) * 2), vbase);
    }
  };

  // Resident Q A-fragments (2 chunks over H=64)
  const bf16* qr = q + (long)(b * T_ + q0 + n) * H_;
  const v16bf aq0 = ld16(qr + g * 8,      qr + 16 + g * 8);
  const v16bf aq1 = ld16(qr + 32 + g * 8, qr + 48 + g * 8);

  v8f o[4] = {};
  float mrow[8], lrow[8];
#pragma unroll
  for (int r = 0; r < 8; ++r) { mrow[r] = -1e30f; lrow[r] = 0.f; }

  bf16* pl = &Pl[wave][0];
  const int nblk = q0b / 32 + 2;                   // covers q0b+63 causally
  issue(0, 0);

  for (int i = 0; i < nblk; ++i) {
    const int k0 = i * 32;
    wait_async0();
    __syncthreads();
    if (i + 1 < nblk) issue(k0 + 32, (i + 1) & 1);

    if (k0 <= q0 + 15) {                           // wave-uniform causal skip
      const bf16* kl = &Kl[i & 1][0];
      const bf16* vl = &Vl[i & 1][0];

      // ---- S = Q K^T : two 16x16 fp32 tiles, all frags loaded up-front --
      const bf16* kr0 = kl + n * 64 + g * 8;
      const bf16* kr1 = kl + (16 + n) * 64 + g * 8;
      v16bf kb00 = ld16(kr0, kr0 + 16);            // tile0, h chunk 0
      v16bf kb10 = ld16(kr1, kr1 + 16);            // tile1, h chunk 0
      v16bf kb01 = ld16(kr0 + 32, kr0 + 48);       // tile0, h chunk 1
      v16bf kb11 = ld16(kr1 + 32, kr1 + 48);       // tile1, h chunk 1
      v8f s0 = {}, s1 = {};
      s0 = WMMA_BF16(aq0, kb00, s0);
      s1 = WMMA_BF16(aq0, kb10, s1);
      s0 = WMMA_BF16(aq1, kb01, s0);
      s1 = WMMA_BF16(aq1, kb11, s1);

      // ---- scale, causal mask, online softmax ---------------------------
      float p0[8], p1[8], alpha[8];
#pragma unroll
      for (int r = 0; r < 8; ++r) {
        const int qi = q0 + g * 8 + r;
        float a0 = s0[r] * 0.125f;                 // 1/sqrt(64)
        float a1 = s1[r] * 0.125f;
        if (k0 + n > qi)      a0 = -1e30f;
        if (k0 + 16 + n > qi) a1 = -1e30f;
        float rm = fmaxf(a0, a1);
#pragma unroll
        for (int msk = 1; msk < 16; msk <<= 1)
          rm = fmaxf(rm, __shfl_xor(rm, msk, 32));
        const float mn = fmaxf(mrow[r], rm);
        alpha[r] = __expf(mrow[r] - mn);
        mrow[r]  = mn;
        p0[r] = __expf(a0 - mn);
        p1[r] = __expf(a1 - mn);
        float rs = p0[r] + p1[r];
#pragma unroll
        for (int msk = 1; msk < 16; msk <<= 1)
          rs += __shfl_xor(rs, msk, 32);
        lrow[r] = lrow[r] * alpha[r] + rs;
      }
#pragma unroll
      for (int nt = 0; nt < 4; ++nt)
#pragma unroll
        for (int r = 0; r < 8; ++r) o[nt][r] *= alpha[r];

      // ---- P -> LDS (C-layout -> row-major 16x32) -----------------------
#pragma unroll
      for (int r = 0; r < 8; ++r) {
        pl[(g * 8 + r) * 32 + n]      = (bf16)p0[r];
        pl[(g * 8 + r) * 32 + 16 + n] = (bf16)p1[r];
      }
      asm volatile("s_wait_dscnt 0" ::: "memory"); // same-wave DS RAW guard
      const v16bf pa = ld16(pl + n * 32 + g * 8, pl + n * 32 + 16 + g * 8);

      // ---- O += P V : all 4 V fragments loaded before first WMMA --------
      v16bf fv[4];
#pragma unroll
      for (int nt = 0; nt < 4; ++nt) {
        const bf16* vr = vl + (nt * 16 + n) * 32 + g * 8;
        fv[nt] = ld16(vr, vr + 16);
      }
#pragma unroll
      for (int nt = 0; nt < 4; ++nt)
        o[nt] = WMMA_BF16(pa, fv[nt], o[nt]);
    }
  }

  // ---- epilogue: O / l -> out [B,T,H] fp32 ------------------------------
#pragma unroll
  for (int nt = 0; nt < 4; ++nt)
#pragma unroll
    for (int r = 0; r < 8; ++r) {
      long row = (long)(b * T_ + q0 + g * 8 + r);
      out[row * H_ + nt * 16 + n] = o[nt][r] / lrow[r];
    }
}

// ---------------------------------------------------------------------------
// Launch. Inputs in setup_inputs() order: x, Wk, Wq, Wv (all fp32).
// Workspace (~3.4 MB): Wt[3][H][C] bf16 | q bf16 [B,T,H] | k bf16 [B,T,H] |
//                      v^T bf16 [B,H,T]
// ---------------------------------------------------------------------------
extern "C" void kernel_launch(void* const* d_in, const int* in_sizes, int n_in,
                              void* d_out, int out_size, void* d_ws, size_t ws_size,
                              hipStream_t stream) {
  (void)in_sizes; (void)n_in; (void)out_size; (void)ws_size;
  const float* x  = (const float*)d_in[0];
  const float* Wk = (const float*)d_in[1];
  const float* Wq = (const float*)d_in[2];
  const float* Wv = (const float*)d_in[3];
  float* out = (float*)d_out;

  char* ws = (char*)d_ws;
  const size_t WT = (size_t)3 * H_ * C_ * sizeof(bf16);    // 384 KB
  const size_t QK = (size_t)B_ * T_ * H_ * sizeof(bf16);   // 1 MB
  bf16* Wt  = (bf16*)(ws);
  bf16* qb  = (bf16*)(ws + WT);
  bf16* kb  = (bf16*)(ws + WT + QK);
  bf16* vtb = (bf16*)(ws + WT + 2 * QK);

  wt_prep<<<dim3((H_ * C_) / 256), dim3(256), 0, stream>>>(Wq, Wk, Wv, Wt);
  qkv_proj<<<dim3((B_ * T_ / 16) / 4), dim3(128), 0, stream>>>(x, Wt, qb, kb, vtb);
  attn<<<dim3((B_ * T_ / 64)), dim3(128), 0, stream>>>(qb, kb, vtb, out);
}
